// Attention_6150393168649
// MI455X (gfx1250) — matrix operand order repaired and checked by reading, not compiled, not run on hardware
//
#include <hip/hip_runtime.h>

// ---------------------------------------------------------------------------
// Transformer-XL style relative-position attention for MI455X (gfx1250).
// GEMMs and attention run on v_wmma_f32_16x16x32_f16 (wave32, f32 accum).
// Global->LDS staging uses GLOBAL_LOAD_ASYNC_TO_LDS_B128 + s_wait_asynccnt.
// Flash-style online softmax: the 16x4x512x512 score tensor never exists.
// ---------------------------------------------------------------------------

typedef __attribute__((ext_vector_type(16))) _Float16 v16h;
typedef __attribute__((ext_vector_type(8)))  _Float16 v8h;
typedef __attribute__((ext_vector_type(8)))  float    v8f;

#define B_   16
#define T_   512
#define D_   256
#define H_   4
#define HD_  64
#define NROW (B_ * T_)

#define USE_ASYNC_LDS 1

// Per-lane async 16B global -> LDS copy (GV mode). lds_off is the byte offset
// inside this workgroup's LDS allocation (hardware adds LDS_BASE). The LDS
// element pointer is passed as an (unused) asm input so the LDS object
// escapes: alias analysis must then honor the "memory" clobber and keep the
// later ds_load fragment reads of the async-written tiles.
static __device__ __forceinline__ void async_ld16(int lds_off, const void* g,
                                                  void* lds_escape) {
#if USE_ASYNC_LDS
  asm volatile("global_load_async_to_lds_b128 %0, %1, off"
               :: "v"(lds_off), "v"(g), "v"(lds_escape) : "memory");
#else
  *(v8h*)lds_escape = *(const v8h*)g; (void)lds_off;
#endif
}
static __device__ __forceinline__ void async_wait0() {
#if USE_ASYNC_LDS
  asm volatile("s_wait_asynccnt 0x0" ::: "memory");
#endif
}

// A-matrix 16x32 f16 fragment (ISA 7.12.2): lane l -> row M = l%16;
// halves 0..7  <- K = (l/16)*8 + 0..7
// halves 8..15 <- K = 16 + (l/16)*8 + 0..7
static __device__ __forceinline__ v16h wmma_a_load(const _Float16* rowp, int hi) {
  v8h lo = *(const v8h*)(rowp + hi * 8);
  v8h hp = *(const v8h*)(rowp + 16 + hi * 8);
  v16h r;
#pragma unroll
  for (int i = 0; i < 8; ++i) { r[i] = lo[i]; r[8 + i] = hp[i]; }
  return r;
}

static __device__ __forceinline__ v8f wmma32(v16h a, v16h b, v8f c) {
  return __builtin_amdgcn_wmma_f32_16x16x32_f16(false, a, false, b,
                                                (short)0, c, false, false);
}

// ---------------------------------------------------------------------------
// Weight transpose + fp32->fp16: WT[k][n] = (f16) W[n][k]   (256x256)
// ---------------------------------------------------------------------------
__global__ __launch_bounds__(256) void wtrans_f16(const float* __restrict__ W,
                                                  _Float16* __restrict__ WT) {
  __shared__ float tile[32][33];
  const int tx = threadIdx.x & 31, ty = threadIdx.x >> 5;
  const int kb = blockIdx.x * 32, nb = blockIdx.y * 32;
#pragma unroll
  for (int r = 0; r < 4; ++r)
    tile[ty + r * 8][tx] = W[(size_t)(nb + ty + r * 8) * D_ + kb + tx];
  __syncthreads();
#pragma unroll
  for (int r = 0; r < 4; ++r)
    WT[(size_t)(kb + ty + r * 8) * D_ + nb + tx] = (_Float16)tile[tx][ty + r * 8];
}

// ---------------------------------------------------------------------------
// Sinusoidal table over 61 clipped relative distances (padded to 64 rows).
// ---------------------------------------------------------------------------
__global__ void posenc_f16(_Float16* __restrict__ ench) {
  const int p = blockIdx.x;   // 0..63 (rows 61..63 zero-padded)
  const int i = threadIdx.x;  // 0..127
  float sv = 0.0f, cv = 0.0f;
  if (p < 61) {
    float div = __expf(-9.210340371976184f * (float)(2 * i) * (1.0f / 256.0f));
    float ang = (float)p * div;
    sv = __sinf(ang);
    cv = __cosf(ang);
  }
  ench[(size_t)p * D_ + 2 * i]     = (_Float16)sv;
  ench[(size_t)p * D_ + 2 * i + 1] = (_Float16)cv;
}

// ---------------------------------------------------------------------------
// LayerNorm over d=256, f16 output. One row per wave32 (8 rows / block).
// ---------------------------------------------------------------------------
__global__ __launch_bounds__(256) void lnorm_f16(const float* __restrict__ x,
                                                 const float* __restrict__ g,
                                                 const float* __restrict__ be,
                                                 _Float16* __restrict__ xh) {
  const int row  = blockIdx.x * 8 + (threadIdx.x >> 5);
  const int lane = threadIdx.x & 31;
  const float4* xr = (const float4*)(x + (size_t)row * D_);
  float4 v0 = xr[lane];
  float4 v1 = xr[lane + 32];

  float s = v0.x + v0.y + v0.z + v0.w + v1.x + v1.y + v1.z + v1.w;
#pragma unroll
  for (int m = 1; m < 32; m <<= 1) s += __shfl_xor(s, m, 32);
  const float mean = s * (1.0f / 256.0f);

  float q = 0.0f;
  q += (v0.x - mean) * (v0.x - mean); q += (v0.y - mean) * (v0.y - mean);
  q += (v0.z - mean) * (v0.z - mean); q += (v0.w - mean) * (v0.w - mean);
  q += (v1.x - mean) * (v1.x - mean); q += (v1.y - mean) * (v1.y - mean);
  q += (v1.z - mean) * (v1.z - mean); q += (v1.w - mean) * (v1.w - mean);
#pragma unroll
  for (int m = 1; m < 32; m <<= 1) q += __shfl_xor(q, m, 32);
  const float rs = rsqrtf(q * (1.0f / 256.0f) + 1e-5f);

  const int c0 = lane * 4, c1 = (lane + 32) * 4;
  _Float16* o = xh + (size_t)row * D_;
  o[c0 + 0] = (_Float16)((v0.x - mean) * rs * g[c0 + 0] + be[c0 + 0]);
  o[c0 + 1] = (_Float16)((v0.y - mean) * rs * g[c0 + 1] + be[c0 + 1]);
  o[c0 + 2] = (_Float16)((v0.z - mean) * rs * g[c0 + 2] + be[c0 + 2]);
  o[c0 + 3] = (_Float16)((v0.w - mean) * rs * g[c0 + 3] + be[c0 + 3]);
  o[c1 + 0] = (_Float16)((v1.x - mean) * rs * g[c1 + 0] + be[c1 + 0]);
  o[c1 + 1] = (_Float16)((v1.y - mean) * rs * g[c1 + 1] + be[c1 + 1]);
  o[c1 + 2] = (_Float16)((v1.z - mean) * rs * g[c1 + 2] + be[c1 + 2]);
  o[c1 + 3] = (_Float16)((v1.w - mean) * rs * g[c1 + 3] + be[c1 + 3]);
}

// ---------------------------------------------------------------------------
// Y[M,256] = A[M,256](f16) x WT[256,256](f16 pre-transposed) + bias.
// One 16x32 strip per wave (two accumulators share each A fragment ->
// 16 WMMAs per wave, 2 per A-frag load). 8 waves cover all 256 columns.
// grid = (M/16).  A tile staged via async global->LDS copies.
// ---------------------------------------------------------------------------
__global__ __launch_bounds__(256) void gemm_nt(
    const _Float16* __restrict__ A, const _Float16* __restrict__ WT,
    const float* __restrict__ bias,
    _Float16* __restrict__ out1, const float* __restrict__ add1,
    _Float16* __restrict__ out2, const float* __restrict__ add2,
    float* __restrict__ outf, _Float16* __restrict__ outT) {
  __shared__ __align__(128) _Float16 As[16][272];  // 16x256 (+16 pad halves)

  const int mt   = blockIdx.x;
  const int wave = threadIdx.x >> 5, lane = threadIdx.x & 31;
  const int l16  = lane & 15, hi = lane >> 4;
  const int n0   = wave * 32;

  // stage the 16x256 A tile: 512 async 16B transfers across 256 threads
  for (int idx = threadIdx.x; idx < 512; idx += 256) {
    int r = idx >> 5, c = (idx & 31) * 8;
    async_ld16((r * 272 + c) * 2,
               A + (size_t)(mt * 16 + r) * D_ + c, &As[r][c]);
  }
  async_wait0();
  __syncthreads();

  v8f acc0 = {}, acc1 = {};
#pragma unroll
  for (int kk = 0; kk < 8; ++kk) {
    v16h a = wmma_a_load(&As[l16][kk * 32], hi);
    // B fragments: lane%16 <-> N column, 16 halves <-> K rows (table order)
    v16h b0, b1;
#pragma unroll
    for (int i = 0; i < 16; ++i) {
      const int kr = kk * 32 + (i < 8 ? hi * 8 + i : 16 + hi * 8 + (i - 8));
      b0[i] = WT[(size_t)kr * D_ + n0 + l16];
      b1[i] = WT[(size_t)kr * D_ + n0 + 16 + l16];
    }
    acc0 = wmma32(a, b0, acc0);
    acc1 = wmma32(a, b1, acc1);
  }

#pragma unroll
  for (int half = 0; half < 2; ++half) {
    const v8f& acc = half ? acc1 : acc0;
    const int col = n0 + half * 16 + l16;
    const float bcol = bias ? bias[col] : 0.0f;
    const float a1 = (out1 && add1) ? add1[col] : 0.0f;
    const float a2 = (out2 && add2) ? add2[col] : 0.0f;
#pragma unroll
    for (int i = 0; i < 8; ++i) {  // C layout: row = i + 8*hi, col
      const int row = mt * 16 + i + 8 * hi;
      const float v = acc[i] + bcol;
      if (out1) out1[(size_t)row * D_ + col] = (_Float16)(v + a1);
      if (out2) out2[(size_t)row * D_ + col] = (_Float16)(v + a2);
      if (outf) outf[(size_t)row * D_ + col] = v;
      if (outT) outT[(size_t)col * 64 + row] = (_Float16)v;  // pos61t[n][r]
    }
  }
}

// ---------------------------------------------------------------------------
// Flash attention with relative-position scores.
// Block = 4 waves (128 thr); grid = (b*h, t/64). Each wave owns 16 query rows.
// ---------------------------------------------------------------------------
struct AttnSmem {
  _Float16 Ks[64][80];       // K chunk transposed: [hd][s]
  _Float16 Vs[64][80];       // V chunk: [s][hd]  (async-staged)
  float    P61[4][16][64];   // per-wave rel-pos dot products
  _Float16 Pex[4][16][80];   // per-wave probability staging
};

__global__ __launch_bounds__(128) void attn_relpos(
    const _Float16* __restrict__ qu, const _Float16* __restrict__ qv,
    const _Float16* __restrict__ kf, const _Float16* __restrict__ vf,
    const _Float16* __restrict__ post, _Float16* __restrict__ ctxh) {
  __shared__ __align__(128) AttnSmem sm;

  const int b = blockIdx.x / H_, h = blockIdx.x % H_;
  const int wave = threadIdx.x >> 5, lane = threadIdx.x & 31;
  const int l16 = lane & 15, hi = lane >> 4;
  const int t0 = blockIdx.y * 64 + wave * 16;

  const _Float16* qub = qu + (size_t)(b * T_ + t0 + l16) * D_ + h * HD_;
  const _Float16* qvb = qv + (size_t)(b * T_ + t0 + l16) * D_ + h * HD_;
  v16h qa0 = wmma_a_load(qub, hi);        // K(hd) = 0..31
  v16h qa1 = wmma_a_load(qub + 32, hi);   // K(hd) = 32..63

  // --- p61[t_local][r] = (q+v_bias) . pos61t  (16x64 tile via 8 WMMAs) ---
  {
    v8f pa[4] = {};
#pragma unroll
    for (int fk = 0; fk < 2; ++fk) {
      v16h a = wmma_a_load(qvb + fk * 32, hi);
#pragma unroll
      for (int j = 0; j < 4; ++j) {
        v16h bfr;
#pragma unroll
        for (int i = 0; i < 16; ++i) {
          const int kr = fk * 32 + (i < 8 ? hi * 8 + i : 16 + hi * 8 + (i - 8));
          bfr[i] = post[(size_t)(h * HD_ + kr) * 64 + j * 16 + l16];
        }
        pa[j] = wmma32(a, bfr, pa[j]);
      }
    }
#pragma unroll
    for (int j = 0; j < 4; ++j)
#pragma unroll
      for (int i = 0; i < 8; ++i)
        sm.P61[wave][i + 8 * hi][j * 16 + l16] = pa[j][i];
  }

  float mrun[8], lrun[8];
  v8f cacc[4] = {};
#pragma unroll
  for (int i = 0; i < 8; ++i) { mrun[i] = -3.0e38f; lrun[i] = 0.0f; }

  for (int s0 = 0; s0 < T_; s0 += 64) {
    __syncthreads();
    // stage V chunk via async copies, K chunk with a manual transpose
    for (int idx = threadIdx.x; idx < 512; idx += 128) {
      int s = idx >> 3, c = (idx & 7) * 8;
      const size_t src = (size_t)(b * T_ + s0 + s) * D_ + h * HD_ + c;
      async_ld16((int)__builtin_offsetof(AttnSmem, Vs) + (s * 80 + c) * 2,
                 vf + src, &sm.Vs[s][c]);
      v8h d = *(const v8h*)(kf + src);
#pragma unroll
      for (int u = 0; u < 8; ++u) sm.Ks[c + u][s] = d[u];
    }
    async_wait0();
    __syncthreads();

    // --- scores: (q+u_bias) . k^T,  16x64 tile ---
    v8f sacc[4] = {};
#pragma unroll
    for (int fk = 0; fk < 2; ++fk) {
      v16h a = fk ? qa1 : qa0;
#pragma unroll
      for (int j = 0; j < 4; ++j) {
        v16h bfr;
#pragma unroll
        for (int i = 0; i < 16; ++i) {
          const int kr = fk * 32 + (i < 8 ? hi * 8 + i : 16 + hi * 8 + (i - 8));
          bfr[i] = sm.Ks[kr][j * 16 + l16];
        }
        sacc[j] = wmma32(a, bfr, sacc[j]);
      }
    }

    // add relative-position score (gather by clipped distance), scale 1/8
#pragma unroll
    for (int j = 0; j < 4; ++j)
#pragma unroll
      for (int i = 0; i < 8; ++i) {
        const int tl = i + 8 * hi, sl = j * 16 + l16;
        int rel = (s0 + sl) - (t0 + tl);
        rel = rel < -30 ? -30 : (rel > 30 ? 30 : rel);
        sacc[j][i] = (sacc[j][i] + sm.P61[wave][tl][rel + 30]) * 0.125f;
      }

    // --- online softmax (row reductions within 16-lane column groups) ---
#pragma unroll
    for (int i = 0; i < 8; ++i) {
      float m = fmaxf(fmaxf(sacc[0][i], sacc[1][i]), fmaxf(sacc[2][i], sacc[3][i]));
#pragma unroll
      for (int msk = 1; msk < 16; msk <<= 1) m = fmaxf(m, __shfl_xor(m, msk, 32));
      const float mn = fmaxf(mrun[i], m);
      const float corr = __expf(mrun[i] - mn);
      mrun[i] = mn;
      lrun[i] *= corr;
#pragma unroll
      for (int j = 0; j < 4; ++j) cacc[j][i] *= corr;
      float rs = 0.0f;
#pragma unroll
      for (int j = 0; j < 4; ++j) {
        const float p = __expf(sacc[j][i] - mn);
        sacc[j][i] = p;
        rs += p;
      }
#pragma unroll
      for (int msk = 1; msk < 16; msk <<= 1) rs += __shfl_xor(rs, msk, 32);
      lrun[i] += rs;
    }

    // probabilities -> LDS (C layout) -> A fragments, then WMMA with V
#pragma unroll
    for (int j = 0; j < 4; ++j)
#pragma unroll
      for (int i = 0; i < 8; ++i)
        sm.Pex[wave][i + 8 * hi][j * 16 + l16] = (_Float16)sacc[j][i];

#pragma unroll
    for (int fk = 0; fk < 2; ++fk) {
      v16h a = wmma_a_load(&sm.Pex[wave][l16][fk * 32], hi);
#pragma unroll
      for (int j = 0; j < 4; ++j) {
        v16h bfr;
#pragma unroll
        for (int i = 0; i < 16; ++i) {
          const int kr = fk * 32 + (i < 8 ? hi * 8 + i : 16 + hi * 8 + (i - 8));
          bfr[i] = sm.Vs[kr][j * 16 + l16];
        }
        cacc[j] = wmma32(a, bfr, cacc[j]);
      }
    }
  }

  // normalize and store context (f16, head-sliced columns)
#pragma unroll
  for (int i = 0; i < 8; ++i) {
    const float inv = 1.0f / lrun[i];
    const size_t row = (size_t)(b * T_ + t0 + i + 8 * hi);
#pragma unroll
    for (int j = 0; j < 4; ++j)
      ctxh[row * D_ + h * HD_ + j * 16 + l16] = (_Float16)(cacc[j][i] * inv);
  }
}

// ---------------------------------------------------------------------------
extern "C" void kernel_launch(void* const* d_in, const int* in_sizes, int n_in,
                              void* d_out, int out_size, void* d_ws, size_t ws_size,
                              hipStream_t stream) {
  (void)in_sizes; (void)n_in; (void)out_size; (void)ws_size;
  const float* x    = (const float*)d_in[0];
  const float* lng  = (const float*)d_in[1];
  const float* lnb  = (const float*)d_in[2];
  const float* Wq   = (const float*)d_in[3];
  const float* bq   = (const float*)d_in[4];
  const float* Wk   = (const float*)d_in[5];
  const float* bk   = (const float*)d_in[6];
  const float* Wv   = (const float*)d_in[7];
  const float* bv   = (const float*)d_in[8];
  const float* Wpos = (const float*)d_in[9];
  const float* bpos = (const float*)d_in[10];
  const float* Wo   = (const float*)d_in[11];
  const float* bo   = (const float*)d_in[12];
  const float* ub   = (const float*)d_in[13];  // [h*hd] == column index
  const float* vb   = (const float*)d_in[14];

  // workspace layout (~25 MB)
  char* ws = (char*)d_ws;
  const size_t MB = 1ull << 20, KB = 1024;
  _Float16* xh    = (_Float16*)(ws + 0 * MB);
  _Float16* qu    = (_Float16*)(ws + 4 * MB);
  _Float16* qvv   = (_Float16*)(ws + 8 * MB);
  _Float16* kf    = (_Float16*)(ws + 12 * MB);
  _Float16* vf    = (_Float16*)(ws + 16 * MB);
  _Float16* ctxh  = (_Float16*)(ws + 20 * MB);
  _Float16* WTq   = (_Float16*)(ws + 24 * MB);
  _Float16* WTk   = (_Float16*)(ws + 24 * MB + 128 * KB);
  _Float16* WTv   = (_Float16*)(ws + 24 * MB + 256 * KB);
  _Float16* WTp   = (_Float16*)(ws + 24 * MB + 384 * KB);
  _Float16* WTo   = (_Float16*)(ws + 24 * MB + 512 * KB);
  _Float16* ench  = (_Float16*)(ws + 24 * MB + 640 * KB);  // 64x256
  _Float16* post  = (_Float16*)(ws + 24 * MB + 704 * KB);  // pos61t [256][64]

  const dim3 tgrid(8, 8);
  wtrans_f16<<<tgrid, 256, 0, stream>>>(Wq,   WTq);
  wtrans_f16<<<tgrid, 256, 0, stream>>>(Wk,   WTk);
  wtrans_f16<<<tgrid, 256, 0, stream>>>(Wv,   WTv);
  wtrans_f16<<<tgrid, 256, 0, stream>>>(Wpos, WTp);
  wtrans_f16<<<tgrid, 256, 0, stream>>>(Wo,   WTo);
  posenc_f16<<<64, 128, 0, stream>>>(ench);
  lnorm_f16<<<NROW / 8, 256, 0, stream>>>(x, lng, lnb, xh);

  // q projection -> q+u_bias and q+v_bias (f16)
  gemm_nt<<<NROW / 16, 256, 0, stream>>>(xh, WTq, bq, qu, ub, qvv, vb, nullptr, nullptr);
  gemm_nt<<<NROW / 16, 256, 0, stream>>>(xh, WTk, bk, kf, nullptr, nullptr, nullptr, nullptr, nullptr);
  gemm_nt<<<NROW / 16, 256, 0, stream>>>(xh, WTv, bv, vf, nullptr, nullptr, nullptr, nullptr, nullptr);
  // 61(->64) x 256 rel-pos projection, written transposed: post[h*64+hd][r]
  gemm_nt<<<4, 256, 0, stream>>>(ench, WTp, bpos, nullptr, nullptr, nullptr, nullptr, nullptr, post);

  attn_relpos<<<dim3(B_ * H_, T_ / 64), 128, 0, stream>>>(qu, qvv, kf, vf, post, ctxh);

  // output projection, f32 to d_out
  gemm_nt<<<NROW / 16, 256, 0, stream>>>(ctxh, WTo, bo, nullptr, nullptr, nullptr, nullptr, (float*)d_out, nullptr);
}
